// TorchGRUClassifier_48919677501995
// MI455X (gfx1250) — matrix operand order repaired and checked by reading, not compiled, not run on hardware
//
#include <hip/hip_runtime.h>
#include <math.h>

typedef __attribute__((ext_vector_type(16))) _Float16 v16h;
typedef __attribute__((ext_vector_type(8)))  _Float16 v8h;
typedef __attribute__((ext_vector_type(8)))  float    v8f;

#define B_  1024
#define T_  256
#define I_  153
#define IP_ 160     // layer-0 K padded to a multiple of 32
#define H_  128
#define G3_ 384     // 3*H

// f16 weight workspace layout (in halves)
#define WIH0_OFF  0
#define WHH0_OFF  (G3_ * IP_)              // 61440
#define WIH1_OFF  (WHH0_OFF + G3_ * H_)    // 110592
#define WHH1_OFF  (WIH1_OFF + G3_ * H_)    // 159744
#define WS_HALVES (WHH1_OFF + G3_ * H_)    // 208896 halves = 417792 B

#define WMMA_F16(a, b, c) \
  __builtin_amdgcn_wmma_f32_16x16x32_f16(false, (a), false, (b), (short)0, (c), false, false)

__device__ __forceinline__ float sigmoidf_(float x) { return 1.0f / (1.0f + __expf(-x)); }
__device__ __forceinline__ float tanhf_(float x) {
  float e = __expf(-2.0f * x);
  return (1.0f - e) / (1.0f + e);
}

// A fragment (16x32 f16, MxK) from a row-major [16][rs] f16 tile (contiguous k = 16*g + s, matching b_frag).
__device__ __forceinline__ v16h a_frag(const _Float16* buf, int rs, int kbase, int lane) {
  const int g = lane >> 4, m = lane & 15;
  const _Float16* p = buf + m * rs + kbase + g * 16;
  union { v16h v; v8h h[2]; } u;
  u.h[0] = *(const v8h*)(p);
  u.h[1] = *(const v8h*)(p + 8);
  return u.v;
}

// B fragment (32x16 f16, KxN) from row-major weights [N][rs]: B(k,n) = W[n][k].
__device__ __forceinline__ v16h b_frag(const _Float16* wblk, int rs, int kbase, int lane) {
  const int g = lane >> 4, n = lane & 15;
  return *(const v16h*)(wblk + (size_t)n * rs + kbase + g * 16);
}

// ------------- kernel 1: all weights fp32 -> f16 (W_ih0 K-padded 153->160) -------------
__global__ void weights_to_f16(const float* __restrict__ Wih0, const float* __restrict__ Whh0,
                               const float* __restrict__ Wih1, const float* __restrict__ Whh1,
                               _Float16* __restrict__ out) {
  int idx = blockIdx.x * blockDim.x + threadIdx.x;
  if (idx >= WS_HALVES) return;
  if (idx < WHH0_OFF) {
    int n = idx / IP_, k = idx - n * IP_;
    out[idx] = (k < I_) ? (_Float16)Wih0[n * I_ + k] : (_Float16)0.0f;
  } else if (idx < WIH1_OFF) {
    out[idx] = (_Float16)Whh0[idx - WHH0_OFF];
  } else if (idx < WHH1_OFF) {
    out[idx] = (_Float16)Wih1[idx - WIH1_OFF];
  } else {
    out[idx] = (_Float16)Whh1[idx - WHH1_OFF];
  }
}

// ------------- kernel 2: fused 2-layer GRU + head; recurrent weights in registers,
//               W_ih1 in LDS, W_ih0 streamed from L2, h carried in registers -------------
__global__ __launch_bounds__(256)
void gru2_fused(const float* __restrict__ x,
                const _Float16* __restrict__ wf,
                const float* __restrict__ bih0, const float* __restrict__ bhh0,
                const float* __restrict__ bih1, const float* __restrict__ bhh1,
                const float* __restrict__ W1, const float* __restrict__ b1,
                const float* __restrict__ W2, const float* __restrict__ b2,
                float* __restrict__ out)
{
  __shared__ __align__(32) _Float16 sWih1[G3_ * H_];   // 96 KB, layer-1 input weights
  __shared__ __align__(32) _Float16 sX[2][16 * IP_];   // 10 KB, double-buffered x tile
  __shared__ __align__(32) _Float16 sH0[2][16 * H_];   // 8 KB
  __shared__ __align__(32) _Float16 sH1[2][16 * H_];   // 8 KB
  __shared__ float sHid[16 * 64];                      // 4 KB

  const int tid  = threadIdx.x;
  const int lane = tid & 31;
  const int wv   = tid >> 5;          // wave 0..7
  const int b0   = blockIdx.x * 16;   // batch tile base
  const int c0   = wv * 16;           // hidden-column block owned by this wave
  const int nl   = lane & 15;
  const int gq   = lane >> 4;

  // ---- recurrent-weight B-fragments live in registers (24 frags = 192 VGPRs) ----
  v16h Bhh0[12], Bhh1[12];            // [ks*3 + gate]
#pragma unroll
  for (int ks = 0; ks < 4; ++ks)
#pragma unroll
    for (int gt = 0; gt < 3; ++gt) {
      const size_t row = (size_t)(gt * H_ + c0) * H_;
      Bhh0[ks * 3 + gt] = b_frag(wf + WHH0_OFF + row, H_, ks * 32, lane);
      Bhh1[ks * 3 + gt] = b_frag(wf + WHH1_OFF + row, H_, ks * 32, lane);
    }

  // ---- W_ih1 -> LDS (vectorized 16B copies of the f16 image) ----
  {
    const uint4* src = (const uint4*)(wf + WIH1_OFF);
    uint4* dst = (uint4*)sWih1;
    for (int i = tid; i < (G3_ * H_) / 8; i += 256) dst[i] = src[i];
  }

  // ---- precompute loop-invariant x-staging descriptors (10 chunks/thread) ----
  // xinfo[c]: bits[25:0] = flat offset into x for this thread's element at t=0,
  //           bit 31 = in-range (k < 153), bit 30 = k==0 (prefetch leader)
  unsigned xinfo[10];
#pragma unroll
  for (int c = 0; c < 10; ++c) {
    int i = tid + c * 256;
    int m = i / IP_, k = i - m * IP_;
    unsigned off = (unsigned)(((b0 + m) * T_) * I_ + k);
    xinfo[c] = off | (k < I_ ? 0x80000000u : 0u) | (k == 0 ? 0x40000000u : 0u);
  }

  // ---- init h buffers, stage x[t=0] ----
  for (int i = tid; i < 16 * H_; i += 256) {
    sH0[0][i] = (_Float16)0.f; sH1[0][i] = (_Float16)0.f;
  }
#pragma unroll
  for (int c = 0; c < 10; ++c) {
    unsigned v = xinfo[c];
    sX[0][tid + c * 256] = (v >> 31) ? (_Float16)x[v & 0x03ffffffu] : (_Float16)0.0f;
  }

  float bi0[3], bh0[3], bi1[3], bh1[3];
#pragma unroll
  for (int gt = 0; gt < 3; ++gt) {
    int col = gt * H_ + c0 + nl;
    bi0[gt] = bih0[col]; bh0[gt] = bhh0[col];
    bi1[gt] = bih1[col]; bh1[gt] = bhh1[col];
  }
  __syncthreads();

  // h carried in registers across timesteps (same lane<->element mapping every step)
  float h0reg[8], h1reg[8];
#pragma unroll
  for (int e = 0; e < 8; ++e) { h0reg[e] = 0.0f; h1reg[e] = 0.0f; }

  for (int t = 0; t < T_; ++t) {
    const int cur = t & 1, nxt = cur ^ 1;

    // ---- barrier-free staging of x[t+1] into the other buffer ----
    if (t + 1 < T_) {
      const float* xt = x + (size_t)(t + 1) * I_;
#pragma unroll
      for (int c = 0; c < 10; ++c) {
        unsigned v = xinfo[c];
        sX[nxt][tid + c * 256] = (v >> 31) ? (_Float16)xt[v & 0x03ffffffu] : (_Float16)0.0f;
        if ((v & 0x40000000u) && t + 2 < T_) __builtin_prefetch(xt + (v & 0x03ffffffu) + I_, 0, 0);
      }
    }

    // ================= layer 0 =================
    v8f gi[3], gh[3];
#pragma unroll
    for (int gt = 0; gt < 3; ++gt)
#pragma unroll
      for (int e = 0; e < 8; ++e) { gi[gt][e] = bi0[gt]; gh[gt][e] = bh0[gt]; }

    // gi0 = x_t @ Wih0^T : K = 160, B-tiles streamed from L2-resident f16 copy
#pragma unroll
    for (int ks = 0; ks < IP_ / 32; ++ks) {
      v16h a = a_frag(&sX[cur][0], IP_, ks * 32, lane);
#pragma unroll
      for (int gt = 0; gt < 3; ++gt) {
        v16h b = b_frag(wf + WIH0_OFF + (size_t)(gt * H_ + c0) * IP_, IP_, ks * 32, lane);
        gi[gt] = WMMA_F16(a, b, gi[gt]);
      }
    }
    // gh0 = h0 @ Whh0^T : K = 128, B in registers
#pragma unroll
    for (int ks = 0; ks < H_ / 32; ++ks) {
      v16h a = a_frag(&sH0[cur][0], H_, ks * 32, lane);
#pragma unroll
      for (int gt = 0; gt < 3; ++gt)
        gh[gt] = WMMA_F16(a, Bhh0[ks * 3 + gt], gh[gt]);
    }
    // GRU elementwise; h_prev comes from the register copy (no LDS read)
#pragma unroll
    for (int e = 0; e < 8; ++e) {
      float hp = h0reg[e];
      float r = sigmoidf_(gi[0][e] + gh[0][e]);
      float z = sigmoidf_(gi[1][e] + gh[1][e]);
      float n = tanhf_(gi[2][e] + r * gh[2][e]);
      h0reg[e] = (1.0f - z) * n + z * hp;
    }
#pragma unroll
    for (int e = 0; e < 8; ++e)
      sH0[nxt][(e + 8 * gq) * H_ + c0 + nl] = (_Float16)h0reg[e];
    __syncthreads();            // B1: h0[nxt] visible for layer 1

    // ================= layer 1 (input = new h0) =================
#pragma unroll
    for (int gt = 0; gt < 3; ++gt)
#pragma unroll
      for (int e = 0; e < 8; ++e) { gi[gt][e] = bi1[gt]; gh[gt][e] = bh1[gt]; }

#pragma unroll
    for (int ks = 0; ks < H_ / 32; ++ks) {
      v16h a  = a_frag(&sH0[nxt][0], H_, ks * 32, lane);
      v16h ah = a_frag(&sH1[cur][0], H_, ks * 32, lane);
#pragma unroll
      for (int gt = 0; gt < 3; ++gt) {
        v16h b = b_frag(sWih1 + (gt * H_ + c0) * H_, H_, ks * 32, lane);
        gi[gt] = WMMA_F16(a,  b,                  gi[gt]);
        gh[gt] = WMMA_F16(ah, Bhh1[ks * 3 + gt],  gh[gt]);
      }
    }
#pragma unroll
    for (int e = 0; e < 8; ++e) {
      float hp = h1reg[e];
      float r = sigmoidf_(gi[0][e] + gh[0][e]);
      float z = sigmoidf_(gi[1][e] + gh[1][e]);
      float n = tanhf_(gi[2][e] + r * gh[2][e]);
      h1reg[e] = (1.0f - z) * n + z * hp;
    }
#pragma unroll
    for (int e = 0; e < 8; ++e)
      sH1[nxt][(e + 8 * gq) * H_ + c0 + nl] = (_Float16)h1reg[e];
    __syncthreads();            // B2: h1[nxt] + sX[nxt] visible for next step
  }

  // ================= head: Linear(128,64)+ReLU -> Linear(64,1)+Sigmoid =================
  const _Float16* hfin = &sH1[T_ & 1][0];   // T_ even -> final h1 is in buffer 0
  for (int p = tid; p < 16 * 64; p += 256) {
    int m = p >> 6, j = p & 63;
    float acc = b1[j];
    for (int k = 0; k < H_; ++k) acc += (float)hfin[m * H_ + k] * W1[j * H_ + k];
    sHid[p] = fmaxf(acc, 0.0f);
  }
  __syncthreads();
  if (tid < 16) {
    float acc = b2[0];
    for (int j = 0; j < 64; ++j) acc += sHid[tid * 64 + j] * W2[j];
    out[b0 + tid] = sigmoidf_(acc);
  }
}

extern "C" void kernel_launch(void* const* d_in, const int* in_sizes, int n_in,
                              void* d_out, int out_size, void* d_ws, size_t ws_size,
                              hipStream_t stream) {
  (void)in_sizes; (void)n_in; (void)out_size; (void)ws_size;
  const float* x    = (const float*)d_in[0];
  const float* Wih0 = (const float*)d_in[1];
  const float* Whh0 = (const float*)d_in[2];
  const float* bih0 = (const float*)d_in[3];
  const float* bhh0 = (const float*)d_in[4];
  const float* Wih1 = (const float*)d_in[5];
  const float* Whh1 = (const float*)d_in[6];
  const float* bih1 = (const float*)d_in[7];
  const float* bhh1 = (const float*)d_in[8];
  const float* W1   = (const float*)d_in[9];
  const float* b1   = (const float*)d_in[10];
  const float* W2   = (const float*)d_in[11];
  const float* b2   = (const float*)d_in[12];

  _Float16* wf = (_Float16*)d_ws;   // 417792 B of f16 weights, L2-resident

  weights_to_f16<<<(WS_HALVES + 255) / 256, 256, 0, stream>>>(Wih0, Whh0, Wih1, Whh1, wf);
  gru2_fused<<<B_ / 16, 256, 0, stream>>>(x, wf, bih0, bhh0, bih1, bhh1,
                                          W1, b1, W2, b2, (float*)d_out);
}
